// HumanVAttention_35253091566005
// MI455X (gfx1250) — matrix operand order repaired and checked by reading, not compiled, not run on hardware
//
#include <hip/hip_runtime.h>
#include <hip/hip_bf16.h>

// ---------------------------------------------------------------------------
// Types for CDNA5 WMMA (wave32): 16x16x32 f16 -> f32
// ---------------------------------------------------------------------------
typedef _Float16 v16h __attribute__((ext_vector_type(16)));
typedef _Float16 v8h  __attribute__((ext_vector_type(8)));
typedef float    v8f  __attribute__((ext_vector_type(8)));

#define T_SEQ   2048
#define HID     2048
#define NH      16
#define NKVH    4
#define DH      128
#define HD      (NH * DH)      // 2048
#define KVD     (NKVH * DH)    // 512
#define BLK_    128
#define NBLK    (T_SEQ / BLK_) // 16
#define LLOC    8
#define GGLB    4
#define GSTRIDE 4
#define NEGV    (-1.0e9f)
#define SCALING 0.08838834764831844f   // 128^-0.5

static __device__ __forceinline__ v8f zero8f() {
    v8f z;
#pragma unroll
    for (int i = 0; i < 8; ++i) z[i] = 0.0f;
    return z;
}

// Build a 16xf16 fragment from two contiguous 8xf16 (16B-aligned) chunks.
static __device__ __forceinline__ v16h ld16(const _Float16* lo, const _Float16* hi) {
    v8h a = *(const v8h*)lo;
    v8h b = *(const v8h*)hi;
    v16h r;
#pragma unroll
    for (int i = 0; i < 8; ++i) { r[i] = a[i]; r[8 + i] = b[i]; }
    return r;
}

static __device__ __forceinline__ v8f wmma_f16(v16h a, v16h b, v8f c) {
    // D = A(16x32 f16) * B(32x16 f16) + C(16x16 f32)
    return __builtin_amdgcn_wmma_f32_16x16x32_f16(
        /*neg_a=*/false, a, /*neg_b=*/false, b,
        /*c_mod=*/(short)0, c, /*reuse_a=*/false, /*reuse_b=*/false);
}

// ---------------------------------------------------------------------------
// f32 -> f16 conversion (grid-stride)
// ---------------------------------------------------------------------------
__global__ void cvt_f32_to_f16(const float* __restrict__ src,
                               _Float16* __restrict__ dst, int n) {
    for (int i = blockIdx.x * blockDim.x + threadIdx.x; i < n;
         i += gridDim.x * blockDim.x)
        dst[i] = (_Float16)src[i];
}

// ---------------------------------------------------------------------------
// Generic WMMA GEMM: C[M,N] = A[M,K] * B[K,N], f16 in, f32 acc, OutT out.
// 128x128 tile per workgroup, 4 waves of 64x64, K-step 32.
// B tile staged transposed in LDS so B-fragments are contiguous along K.
// ---------------------------------------------------------------------------
template <typename OutT>
__global__ __launch_bounds__(128) void gemm_f16_wmma(
    const _Float16* __restrict__ A, const _Float16* __restrict__ B,
    OutT* __restrict__ C, int M, int N, int Kdim) {
    __shared__ _Float16 Bst[128][40];   // [colLocal][k], padded stride (80B = 5*16B)

    const int tid  = threadIdx.x;
    const int w    = tid >> 5;
    const int lane = tid & 31;
    const int l16  = lane & 15;
    const int hi   = lane >> 4;
    const int koffA = hi ? 8 : 0;     // A-fragment K offset (16-bit A layout)
    const int koffB = hi ? 8 : 0;     // B-fragment K offset

    const int tileN = blockIdx.x * 128;
    const int tileM = blockIdx.y * 128;
    const int wr = w >> 1, wc = w & 1;
    const int m0 = tileM + wr * 64;
    const int n0 = tileN + wc * 64;

    v8f acc[4][4];
#pragma unroll
    for (int i = 0; i < 4; ++i)
#pragma unroll
        for (int j = 0; j < 4; ++j) acc[i][j] = zero8f();

    for (int kt = 0; kt < Kdim; kt += 32) {
        // Cooperative transposed stage of the 32xK by 128-wide B tile.
        for (int e = tid; e < 32 * 128; e += 128) {
            int k = e >> 7, c = e & 127;
            Bst[c][k] = B[(size_t)(kt + k) * N + tileN + c];
        }
        __syncthreads();

        v16h af[4];
#pragma unroll
        for (int i = 0; i < 4; ++i) {
            const _Float16* ar = A + (size_t)(m0 + 16 * i + l16) * Kdim + kt + koffA;
            af[i] = ld16(ar, ar + 16);
        }
#pragma unroll
        for (int j = 0; j < 4; ++j) {
            const _Float16* br = &Bst[wc * 64 + 16 * j + l16][koffB];
            v16h bf = ld16(br, br + 16);
#pragma unroll
            for (int i = 0; i < 4; ++i) acc[i][j] = wmma_f16(af[i], bf, acc[i][j]);
        }
        __syncthreads();
    }

#pragma unroll
    for (int i = 0; i < 4; ++i)
#pragma unroll
        for (int j = 0; j < 4; ++j)
#pragma unroll
            for (int r = 0; r < 8; ++r) {
                int row = m0 + 16 * i + r + hi * 8;
                int col = n0 + 16 * j + l16;
                C[(size_t)row * N + col] = (OutT)acc[i][j][r];
            }
}

// ---------------------------------------------------------------------------
// RoPE in-place on f16 projections. One thread per (t, head, d<64) pair.
// ---------------------------------------------------------------------------
__global__ void rope_kernel(_Float16* __restrict__ X,
                            const float* __restrict__ cosb,
                            const float* __restrict__ sinb, int heads) {
    int total = T_SEQ * heads * 64;
    int idx = blockIdx.x * blockDim.x + threadIdx.x;
    if (idx >= total) return;
    int t = idx / (heads * 64);
    int r = idx % (heads * 64);
    int h = r / 64, d = r % 64;
    int ld = heads * DH;
    size_t base = (size_t)t * ld + h * DH;
    float x1 = (float)X[base + d];
    float x2 = (float)X[base + d + 64];
    float c1 = cosb[t * DH + d],      s1 = sinb[t * DH + d];
    float c2 = cosb[t * DH + d + 64], s2 = sinb[t * DH + d + 64];
    X[base + d]      = (_Float16)(x1 * c1 - x2 * s1);
    X[base + d + 64] = (_Float16)(x2 * c2 + x1 * s2);
}

// ---------------------------------------------------------------------------
// Per-block mean pooling of K/V (mask is all-ones -> denom = BLK).
// ---------------------------------------------------------------------------
__global__ void pool_kv_kernel(const _Float16* __restrict__ K,
                               const _Float16* __restrict__ V,
                               _Float16* __restrict__ Kp,
                               _Float16* __restrict__ Vp) {
    int idx = blockIdx.x * blockDim.x + threadIdx.x;   // NBLK * KVD
    if (idx >= NBLK * KVD) return;
    int bi = idx / KVD, f = idx % KVD;
    float sk = 0.f, sv = 0.f;
    for (int tkn = 0; tkn < BLK_; ++tkn) {
        size_t o = (size_t)(bi * BLK_ + tkn) * KVD + f;
        sk += (float)K[o];
        sv += (float)V[o];
    }
    Kp[idx] = (_Float16)(sk * (1.0f / BLK_));
    Vp[idx] = (_Float16)(sv * (1.0f / BLK_));
}

// ---------------------------------------------------------------------------
// Fused block-sparse attention. One workgroup per (head, q-block).
// 4 waves, each owning 32 query rows. Streaming (flash) softmax over the
// 8 local key blocks + one 32-slot pooled-global pseudo-block.
// ---------------------------------------------------------------------------
__global__ __launch_bounds__(128) void attn_kernel(
    const _Float16* __restrict__ Q,  const _Float16* __restrict__ Kc,
    const _Float16* __restrict__ Vc, const _Float16* __restrict__ Kp,
    const _Float16* __restrict__ Vp, _Float16* __restrict__ O) {
    const int h   = blockIdx.x;       // head
    const int nb  = blockIdx.y;       // query block
    const int kvh = h >> 2;           // H/KVH == 4
    const int tid  = threadIdx.x;
    const int w    = tid >> 5;
    const int lane = tid & 31;
    const int l16  = lane & 15;
    const int hi   = lane >> 4;
    const int koffA = hi ? 8 : 0;
    const int koffB = hi ? 8 : 0;

    __shared__ _Float16 Ps[4][32][136];  // per-wave prob staging (A-frag friendly)
    __shared__ _Float16 Kg[32][136];     // pooled-global key block (rows 0..3 live)
    __shared__ _Float16 Vg[32][136];

    // ---- fill pooled-global block ----
    const int gpos = ((nb > 0 ? nb - 1 : 0) / GSTRIDE);
    for (int e = tid; e < 32 * DH; e += 128) {
        int row = e >> 7, d = e & 127;
        _Float16 kv = (_Float16)0.0f, vv = (_Float16)0.0f;
        if (row < GGLB) {
            int ig = (gpos - (GGLB - 1 - row)) * GSTRIDE;   // block index
            if (nb > 0 && ig >= 0) {
                kv = Kp[(size_t)ig * KVD + kvh * DH + d];
                vv = Vp[(size_t)ig * KVD + kvh * DH + d];
            }
        }
        Kg[row][d] = kv;
        Vg[row][d] = vv;
    }
    __syncthreads();

    // ---- persistent Q fragments (2 row tiles x 4 K-chunks of 32) ----
    v16h qf[2][4];
#pragma unroll
    for (int i = 0; i < 2; ++i) {
        int tok = nb * BLK_ + w * 32 + 16 * i + l16;
        const _Float16* qrow = Q + (size_t)tok * HD + h * DH;
#pragma unroll
        for (int kd = 0; kd < 4; ++kd)
            qf[i][kd] = ld16(qrow + kd * 32 + koffA, qrow + kd * 32 + 16 + koffA);
    }

    float mrun[2][8], lrun[2][8];
    v8f accO[2][8];
#pragma unroll
    for (int i = 0; i < 2; ++i)
#pragma unroll
        for (int r = 0; r < 8; ++r) { mrun[i][r] = -1.0e30f; lrun[i][r] = 0.0f; }
#pragma unroll
    for (int i = 0; i < 2; ++i)
#pragma unroll
        for (int j = 0; j < 8; ++j) accO[i][j] = zero8f();

    const int rowInBlkBase = w * 32 + hi * 8;   // + 16*i + r -> query row in block

    // ================= local sliding-window blocks =================
    for (int lb = 0; lb < LLOC; ++lb) {
        int kb = nb - (LLOC - 1) + lb;
        if (kb < 0) continue;
        const bool causal = (kb == nb);
        const _Float16* Kbase = Kc + (size_t)kb * BLK_ * KVD + kvh * DH;
        const _Float16* Vbase = Vc + (size_t)kb * BLK_ * KVD + kvh * DH;
        if (kb < nb)   // prefetch next key block into caches (global_prefetch_b8)
            __builtin_prefetch(Kbase + (size_t)BLK_ * KVD + lane * 64, 0, 1);

        for (int i = 0; i < 2; ++i) {
            // ---- S = Q * K^T for this 16-row tile ----
            v8f s[8];
#pragma unroll
            for (int j = 0; j < 8; ++j) s[j] = zero8f();
#pragma unroll
            for (int kd = 0; kd < 4; ++kd) {
#pragma unroll
                for (int j = 0; j < 8; ++j) {
                    const _Float16* kr =
                        Kbase + (size_t)(16 * j + l16) * KVD + kd * 32 + koffB;
                    v16h bf = ld16(kr, kr + 16);
                    s[j] = wmma_f16(qf[i][kd], bf, s[j]);
                }
            }
            // ---- scale + causal mask + online softmax update ----
#pragma unroll
            for (int r = 0; r < 8; ++r) {
                int qrow = rowInBlkBase + 16 * i + r;
                float mx = mrun[i][r];
#pragma unroll
                for (int j = 0; j < 8; ++j) {
                    float sv = s[j][r] * SCALING;
                    int col = 16 * j + l16;
                    if (causal && col > qrow) sv = NEGV;
                    s[j][r] = sv;
                    mx = fmaxf(mx, sv);
                }
#pragma unroll
                for (int msk = 1; msk < 16; msk <<= 1)
                    mx = fmaxf(mx, __shfl_xor(mx, msk, 32));
                float factor = __expf(mrun[i][r] - mx);
                float rs = 0.0f;
#pragma unroll
                for (int j = 0; j < 8; ++j) {
                    float p = __expf(s[j][r] - mx);
                    rs += p;
                    Ps[w][16 * i + r + hi * 8][16 * j + l16] = (_Float16)p;
                }
#pragma unroll
                for (int msk = 1; msk < 16; msk <<= 1)
                    rs += __shfl_xor(rs, msk, 32);
                lrun[i][r] = lrun[i][r] * factor + rs;
                mrun[i][r] = mx;
#pragma unroll
                for (int jd = 0; jd < 8; ++jd) accO[i][jd][r] *= factor;
            }
        }
        __syncthreads();   // Ps fully written (also keeps waves in phase)

        // ---- O += P * V ----
        for (int i = 0; i < 2; ++i) {
#pragma unroll
            for (int kc = 0; kc < 4; ++kc) {
                const _Float16* pr = &Ps[w][16 * i + l16][kc * 32 + koffA];
                v16h af = ld16(pr, pr + 16);
#pragma unroll
                for (int jd = 0; jd < 8; ++jd) {
                    v16h bf;
                    const _Float16* vcol =
                        Vbase + (size_t)(kc * 32 + koffB) * KVD + 16 * jd + l16;
#pragma unroll
                    for (int t = 0; t < 16; ++t) bf[t] = vcol[(size_t)(t + (t >= 8 ? 8 : 0)) * KVD];
                    accO[i][jd] = wmma_f16(af, bf, accO[i][jd]);
                }
            }
        }
        __syncthreads();
    }

    // ================= pooled-global pseudo-block (32 slots, 4 valid) =======
    for (int i = 0; i < 2; ++i) {
        v8f s[2];
#pragma unroll
        for (int j = 0; j < 2; ++j) s[j] = zero8f();
#pragma unroll
        for (int kd = 0; kd < 4; ++kd) {
#pragma unroll
            for (int j = 0; j < 2; ++j) {
                const _Float16* kr = &Kg[16 * j + l16][kd * 32 + koffB];
                v16h bf = ld16(kr, kr + 16);
                s[j] = wmma_f16(qf[i][kd], bf, s[j]);
            }
        }
#pragma unroll
        for (int r = 0; r < 8; ++r) {
            float mx = mrun[i][r];
#pragma unroll
            for (int j = 0; j < 2; ++j) {
                int col = 16 * j + l16;
                int ig  = (gpos - (GGLB - 1 - col)) * GSTRIDE;
                bool ok = (col < GGLB) && (nb > 0) && (ig >= 0);
                float sv = ok ? s[j][r] * SCALING : NEGV;
                s[j][r] = sv;
                mx = fmaxf(mx, sv);
            }
#pragma unroll
            for (int msk = 1; msk < 16; msk <<= 1)
                mx = fmaxf(mx, __shfl_xor(mx, msk, 32));
            float factor = __expf(mrun[i][r] - mx);
            float rs = 0.0f;
#pragma unroll
            for (int j = 0; j < 2; ++j) {
                float p = __expf(s[j][r] - mx);
                rs += p;
                Ps[w][16 * i + r + hi * 8][16 * j + l16] = (_Float16)p;
            }
#pragma unroll
            for (int msk = 1; msk < 16; msk <<= 1)
                rs += __shfl_xor(rs, msk, 32);
            lrun[i][r] = lrun[i][r] * factor + rs;
            mrun[i][r] = mx;
#pragma unroll
            for (int jd = 0; jd < 8; ++jd) accO[i][jd][r] *= factor;
        }
    }
    __syncthreads();
    for (int i = 0; i < 2; ++i) {
        const _Float16* pr = &Ps[w][16 * i + l16][koffA];
        v16h af = ld16(pr, pr + 16);
#pragma unroll
        for (int jd = 0; jd < 8; ++jd) {
            v16h bf;
#pragma unroll
            for (int t = 0; t < 16; ++t) bf[t] = Vg[koffB + t + (t >= 8 ? 8 : 0)][16 * jd + l16];
            accO[i][jd] = wmma_f16(af, bf, accO[i][jd]);
        }
    }

    // ================= epilogue: O / l -> f16 =================
#pragma unroll
    for (int i = 0; i < 2; ++i)
#pragma unroll
        for (int r = 0; r < 8; ++r) {
            int tok = nb * BLK_ + rowInBlkBase + 16 * i + r;
            float inv = 1.0f / fmaxf(lrun[i][r], 1.0e-20f);
#pragma unroll
            for (int jd = 0; jd < 8; ++jd)
                O[(size_t)tok * HD + h * DH + 16 * jd + l16] =
                    (_Float16)(accO[i][jd][r] * inv);
        }
}

// ---------------------------------------------------------------------------
// Host-side orchestration
// ---------------------------------------------------------------------------
extern "C" void kernel_launch(void* const* d_in, const int* in_sizes, int n_in,
                              void* d_out, int out_size, void* d_ws, size_t ws_size,
                              hipStream_t stream) {
    (void)in_sizes; (void)n_in; (void)out_size; (void)ws_size;
    const float* hidden = (const float*)d_in[0];
    const float* cosb   = (const float*)d_in[1];
    const float* sinb   = (const float*)d_in[2];
    const float* Wq     = (const float*)d_in[3];
    const float* Wk     = (const float*)d_in[4];
    const float* Wv     = (const float*)d_in[5];
    const float* Wo     = (const float*)d_in[6];
    float* out = (float*)d_out;

    char* ws = (char*)d_ws;
    size_t off = 0;
    auto alloc = [&](size_t bytes) -> void* {
        void* p = ws + off;
        off += (bytes + 255) & ~(size_t)255;
        return p;
    };
    _Float16* Xh   = (_Float16*)alloc((size_t)T_SEQ * HID * 2);
    _Float16* Wqh  = (_Float16*)alloc((size_t)HID * HD * 2);
    _Float16* Wkh  = (_Float16*)alloc((size_t)HID * KVD * 2);
    _Float16* Wvh  = (_Float16*)alloc((size_t)HID * KVD * 2);
    _Float16* Woh  = (_Float16*)alloc((size_t)HD * HID * 2);
    _Float16* Qb   = (_Float16*)alloc((size_t)T_SEQ * HD * 2);
    _Float16* Kb   = (_Float16*)alloc((size_t)T_SEQ * KVD * 2);
    _Float16* Vb   = (_Float16*)alloc((size_t)T_SEQ * KVD * 2);
    _Float16* Kp   = (_Float16*)alloc((size_t)NBLK * KVD * 2);
    _Float16* Vp   = (_Float16*)alloc((size_t)NBLK * KVD * 2);
    _Float16* AOut = (_Float16*)alloc((size_t)T_SEQ * HD * 2);

    // 1) f32 -> f16 conversions
    cvt_f32_to_f16<<<2048, 256, 0, stream>>>(hidden, Xh, T_SEQ * HID);
    cvt_f32_to_f16<<<2048, 256, 0, stream>>>(Wq, Wqh, HID * HD);
    cvt_f32_to_f16<<<1024, 256, 0, stream>>>(Wk, Wkh, HID * KVD);
    cvt_f32_to_f16<<<1024, 256, 0, stream>>>(Wv, Wvh, HID * KVD);
    cvt_f32_to_f16<<<2048, 256, 0, stream>>>(Wo, Woh, HD * HID);

    // 2) QKV projections (WMMA GEMMs)
    gemm_f16_wmma<_Float16><<<dim3(HD / 128, T_SEQ / 128), 128, 0, stream>>>(
        Xh, Wqh, Qb, T_SEQ, HD, HID);
    gemm_f16_wmma<_Float16><<<dim3(KVD / 128, T_SEQ / 128), 128, 0, stream>>>(
        Xh, Wkh, Kb, T_SEQ, KVD, HID);
    gemm_f16_wmma<_Float16><<<dim3(KVD / 128, T_SEQ / 128), 128, 0, stream>>>(
        Xh, Wvh, Vb, T_SEQ, KVD, HID);

    // 3) RoPE (in-place)
    rope_kernel<<<(T_SEQ * NH * 64 + 255) / 256, 256, 0, stream>>>(Qb, cosb, sinb, NH);
    rope_kernel<<<(T_SEQ * NKVH * 64 + 255) / 256, 256, 0, stream>>>(Kb, cosb, sinb, NKVH);

    // 4) pooled global K/V
    pool_kv_kernel<<<(NBLK * KVD + 255) / 256, 256, 0, stream>>>(Kb, Vb, Kp, Vp);

    // 5) fused block-sparse attention
    attn_kernel<<<dim3(NH, NBLK), 128, 0, stream>>>(Qb, Kb, Vb, Kp, Vp, AOut);

    // 6) output projection -> f32 d_out
    gemm_f16_wmma<float><<<dim3(HID / 128, T_SEQ / 128), 128, 0, stream>>>(
        AOut, Woh, out, T_SEQ, HID, HD);
}
